// Attention_67070209294337
// MI455X (gfx1250) — matrix operand order repaired and checked by reading, not compiled, not run on hardware
//
#include <hip/hip_runtime.h>

// ---------------------------------------------------------------------------
// Causal attention with single head projection, CDNA5 (gfx1250) WMMA version.
//   Wswz = W pre-swizzled into per-lane B-fragment layout (f16)   [one-time]
//   Qp = (q @ Wq + bq) * 1/sqrt(64)   -> f16, row-major [B][S][64]
//   Kp = (k @ Wk + bk)                -> f16, row-major [B][S][64]
//   Vt = (v @ Wv + bv)                -> f16, DIM-major [B][64][S] (transposed)
//   out = softmax_causal(Qp Kp^T) Vp  -> f32 [B][S][64]
// All GEMMs via v_wmma_f32_16x16x32_f16. Scores computed transposed
// (S^T = K Q^T) so every WMMA operand is a contiguous vector load and the
// P^T fragment for P@V is built lane-locally + one shfl_xor(16) per reg.
// Projection is software-pipelined with ping-pong fragment buffers
// (unroll-by-2, no register rotation copies).
// ---------------------------------------------------------------------------

typedef __attribute__((ext_vector_type(16))) _Float16 v16h;
typedef __attribute__((ext_vector_type(8)))  _Float16 v8h;
typedef __attribute__((ext_vector_type(8)))  float    v8f;
typedef __attribute__((ext_vector_type(4)))  float    v4f;

#define WMMA_F16(A, B, C) \
  __builtin_amdgcn_wmma_f32_16x16x32_f16(false, (A), false, (B), (short)0, (C), false, false)

#define D_MODEL 1024
#define D_K     64
#define SEQ     2048
#define BATCH   8

// A-matrix 16x32 f16 fragment (lane = M row; half0 K=0..15; half1 K=16..31)
static __device__ inline v16h load_fragA_h(const _Float16* base, int row, int stride,
                                           int kOff, int lane) {
  const int m = lane & 15, h = lane >> 4;
  const _Float16* p = base + (size_t)(row + m) * stride + kOff + h * 16;
  v8h lo = *(const v8h*)p;
  v8h hi = *(const v8h*)(p + 8);
  v16h f;
#pragma unroll
  for (int i = 0; i < 8; ++i) { f[i] = lo[i]; f[8 + i] = hi[i]; }
  return f;
}

// B-matrix 32x16 f16 fragment (lane = N col; half0 K=0..15; half1 K=16..31)
static __device__ inline v16h load_fragB_h(const _Float16* base, int row, int stride,
                                           int kOff, int lane) {
  const int n = lane & 15, h = lane >> 4;
  const _Float16* p = base + (size_t)(row + n) * stride + kOff + h * 16;
  v8h lo = *(const v8h*)p;
  v8h hi = *(const v8h*)(p + 8);
  v16h f;
#pragma unroll
  for (int i = 0; i < 8; ++i) { f[i] = lo[i]; f[8 + i] = hi[i]; }
  return f;
}

// A fragment loaded from f32 source with on-the-fly f16 convert
static __device__ inline v16h load_fragA_f32(const float* base, int row, int stride,
                                             int kOff, int lane) {
  const int m = lane & 15, h = lane >> 4;
  const float* p = base + (size_t)(row + m) * stride + kOff + h * 16;
  v4f a0 = *(const v4f*)(p);
  v4f a1 = *(const v4f*)(p + 4);
  v4f b0 = *(const v4f*)(p + 8);
  v4f b1 = *(const v4f*)(p + 12);
  v16h f;
#pragma unroll
  for (int i = 0; i < 4; ++i) {
    f[i]      = (_Float16)a0[i];
    f[4 + i]  = (_Float16)a1[i];
    f[8 + i]  = (_Float16)b0[i];
    f[12 + i] = (_Float16)b1[i];
  }
  return f;
}

// ---------------------------------------------------------------------------
// Kernel 0: pre-swizzle the three 1024x64 weight matrices into per-lane
// B-fragment order:  Wswz[which][kc][nc][lane][16]  (f16, 32B per lane).
// grid = (32, 3), block = 128 (wave index = nc).
// ---------------------------------------------------------------------------
__global__ __launch_bounds__(128) void wswz_kernel(
    const float* __restrict__ Wq, const float* __restrict__ Wk, const float* __restrict__ Wv,
    _Float16* __restrict__ Wswz)
{
  const int lane = threadIdx.x & 31, nc = threadIdx.x >> 5;
  const int kc = blockIdx.x;          // k-chunk of 32
  const int which = blockIdx.y;       // 0:Q 1:K 2:V
  const float* W = (which == 0) ? Wq : (which == 1) ? Wk : Wv;
  const int n = lane & 15, h = lane >> 4;
  const int kb = kc * 32;
  v16h f;
#pragma unroll
  for (int e = 0; e < 16; ++e)
    f[e] = (_Float16)W[(size_t)(kb + h * 16 + e) * D_K + nc * 16 + n];
  _Float16* dst = Wswz + (((size_t)which * 32 + kc) * 4 + nc) * (32 * 16) + lane * 16;
  *(v16h*)dst = f;
}

// ---------------------------------------------------------------------------
// Kernel 1: fused Q/K/V projection. grid = (256, 3), block = 128 (4 waves).
// wave handles 16 flattened rows (b*S+s); blockIdx.y selects Q/K/V.
// Ping-pong pipelined: while phase-A fragments feed the WMMAs, phase-B
// fragments are loading, and vice versa (unroll-by-2, no rotation movs).
// ---------------------------------------------------------------------------
__global__ __launch_bounds__(128) void proj_kernel(
    const float* __restrict__ q, const float* __restrict__ k, const float* __restrict__ v,
    const float* __restrict__ bq, const float* __restrict__ bk, const float* __restrict__ bv,
    const _Float16* __restrict__ Wswz,
    _Float16* __restrict__ Qp, _Float16* __restrict__ Kp, _Float16* __restrict__ Vt)
{
  const int lane = threadIdx.x & 31, wave = threadIdx.x >> 5;
  const int which = blockIdx.y;
  const float* X    = (which == 0) ? q  : (which == 1) ? k  : v;
  const float* bias = (which == 0) ? bq : (which == 1) ? bk : bv;
  const _Float16* Wf = Wswz + (size_t)which * 32 * 4 * (32 * 16) + lane * 16;
  const int rb = blockIdx.x * 64 + wave * 16;

  v8f acc[4] = {};

  // prologue: phase-A fragments for kc = 0
  v16h aA = load_fragA_f32(X, rb, D_MODEL, 0, lane);
  v16h wA[4];
#pragma unroll
  for (int nc = 0; nc < 4; ++nc)
    wA[nc] = *(const v16h*)(Wf + (size_t)nc * (32 * 16));

  for (int kc = 0; kc < 32; kc += 2) {
    // load phase-B (kc+1) while computing phase-A (kc)
    v16h aB = load_fragA_f32(X, rb, D_MODEL, (kc + 1) * 32, lane);
    v16h wB[4];
    {
      const _Float16* wp = Wf + (size_t)(kc + 1) * 4 * (32 * 16);
#pragma unroll
      for (int nc = 0; nc < 4; ++nc)
        wB[nc] = *(const v16h*)(wp + (size_t)nc * (32 * 16));
    }
#pragma unroll
    for (int nc = 0; nc < 4; ++nc)
      acc[nc] = WMMA_F16(aA, wA[nc], acc[nc]);

    // load phase-A (kc+2, clamped) while computing phase-B (kc+1)
    const int kcn = (kc + 2 < 32) ? kc + 2 : 31;
    aA = load_fragA_f32(X, rb, D_MODEL, kcn * 32, lane);
    {
      const _Float16* wp = Wf + (size_t)kcn * 4 * (32 * 16);
#pragma unroll
      for (int nc = 0; nc < 4; ++nc)
        wA[nc] = *(const v16h*)(wp + (size_t)nc * (32 * 16));
    }
#pragma unroll
    for (int nc = 0; nc < 4; ++nc)
      acc[nc] = WMMA_F16(aB, wB[nc], acc[nc]);
  }

  const int n = lane & 15, h = lane >> 4;
  const float scale = (which == 0) ? 0.125f : 1.0f;  // fold 1/sqrt(d_k) into Q

  if (which < 2) {
    // row-major [b*S+s][64] f16
    _Float16* out = (which == 0) ? Qp : Kp;
#pragma unroll
    for (int nc = 0; nc < 4; ++nc) {
      const int col = nc * 16 + n;
      const float bb = bias[col];
#pragma unroll
      for (int r = 0; r < 8; ++r) {
        const int row = rb + r + 8 * h;
        out[(size_t)row * D_K + col] = (_Float16)((acc[nc][r] + bb) * scale);
      }
    }
  } else {
    // V stored dim-major: Vt[b][d][s]; regs r are consecutive s -> one b128 store
    const int bidx = rb >> 11;                 // rb / 2048
    const int s0   = (rb & 2047) + 8 * h;
#pragma unroll
    for (int nc = 0; nc < 4; ++nc) {
      const int col = nc * 16 + n;
      const float bb = bias[col];
      v8h pk;
#pragma unroll
      for (int r = 0; r < 8; ++r) pk[r] = (_Float16)(acc[nc][r] + bb);
      *(v8h*)(Vt + ((size_t)(bidx * D_K + col)) * SEQ + s0) = pk;
    }
  }
}

// ---------------------------------------------------------------------------
// Kernel 2: flash attention (transposed-score form). grid = (32, 8), block=128.
// wave owns 16 query rows; inner loop walks 32 keys/step (causal bound).
// K/V per batch is 512KB -> L2-resident; all 8 fragment loads of a step are
// issued up front so QK^T WMMAs + softmax VALU cover the V-load latency.
// ---------------------------------------------------------------------------
__global__ __launch_bounds__(128) void attn_kernel(
    const _Float16* __restrict__ Qp, const _Float16* __restrict__ Kp,
    const _Float16* __restrict__ Vt, float* __restrict__ out)
{
  const int lane = threadIdx.x & 31, wave = threadIdx.x >> 5;
  const int n = lane & 15, h = lane >> 4;
  const int b = blockIdx.y;
  const int qb = blockIdx.x * 64 + wave * 16;
  const _Float16* Qb = Qp + (size_t)b * SEQ * D_K;
  const _Float16* Kb = Kp + (size_t)b * SEQ * D_K;
  const _Float16* Vb = Vt + (size_t)b * D_K * SEQ;
  const int qrow = qb + n;   // this lane's query (both halves agree)

  // Q^T B-fragments (reused across whole key loop), dims 0..31 and 32..63
  v16h bq0 = load_fragB_h(Qb, qb, D_K, 0, lane);
  v16h bq1 = load_fragB_h(Qb, qb, D_K, 32, lane);

  v8f acc[4] = {};                 // O^T accum: [dim chunk][reg]=dim, lane=query
  float mrow = -3.0e38f, lrow = 0.0f;
  const int nIter = (qb + 16 + 31) >> 5;

  for (int it = 0; it < nIter; ++it) {
    const int kb = it * 32;

    // prefetch next key-step's K rows + V^T columns (speculative, L2-resident)
    __builtin_prefetch(Kb + (size_t)(kb + 32 + lane) * D_K, 0, 1);
    __builtin_prefetch(Vb + (size_t)lane * SEQ + kb + 32, 0, 1);
    __builtin_prefetch(Vb + (size_t)(lane + 32) * SEQ + kb + 32, 0, 1);

    // issue ALL of this step's fragment loads up front (K then V)
    v16h ak0 = load_fragA_h(Kb, kb,      D_K, 0,  lane);
    v16h ak1 = load_fragA_h(Kb, kb,      D_K, 32, lane);
    v16h ak2 = load_fragA_h(Kb, kb + 16, D_K, 0,  lane);
    v16h ak3 = load_fragA_h(Kb, kb + 16, D_K, 32, lane);
    v16h av[4];
#pragma unroll
    for (int dc = 0; dc < 4; ++dc)
      av[dc] = load_fragA_h(Vb, dc * 16, SEQ, kb, lane);

    // S^T tiles: s0 = keys kb..kb+15, s1 = keys kb+16..kb+31 (x 16 queries)
    v8f s0 = {}, s1 = {};
    s0 = WMMA_F16(ak0, bq0, s0);
    s0 = WMMA_F16(ak1, bq1, s0);
    s1 = WMMA_F16(ak2, bq0, s1);
    s1 = WMMA_F16(ak3, bq1, s1);

    // causal mask + running max (C layout: reg r holds key kb(+16) + r + 8h)
    float e0[8], e1[8];
    float mloc = -3.0e38f;
#pragma unroll
    for (int r = 0; r < 8; ++r) {
      const int k0 = kb + 8 * h + r;
      const int k1 = kb + 16 + 8 * h + r;
      e0[r] = (k0 <= qrow) ? s0[r] : -3.0e38f;
      e1[r] = (k1 <= qrow) ? s1[r] : -3.0e38f;
      mloc = fmaxf(mloc, fmaxf(e0[r], e1[r]));
    }
    mloc = fmaxf(mloc, __shfl_xor(mloc, 16, 32));  // combine wave halves
    const float mnew  = fmaxf(mrow, mloc);
    const float alpha = __expf(mrow - mnew);
    mrow = mnew;

    float ssum = 0.0f;
#pragma unroll
    for (int r = 0; r < 8; ++r) {
      e0[r] = __expf(e0[r] - mnew);
      e1[r] = __expf(e1[r] - mnew);
      ssum += e0[r] + e1[r];
    }
    ssum += __shfl_xor(ssum, 16, 32);
    lrow = lrow * alpha + ssum;

    // Build P^T B-fragment. Each half needs only the partner's OTHER chunk:
    // h=0 needs partner e0 (chunk0 rows 8..15); h=1 needs partner e1 (chunk1
    // rows 0..7). One symmetric shuffle per reg: send what the partner needs.
    v16h pb;
#pragma unroll
    for (int r = 0; r < 8; ++r) {
      const float send = h ? e0[r] : e1[r];
      const float recv = __shfl_xor(send, 16, 32);
      pb[r]     = (_Float16)(h ? recv  : e0[r]);
      pb[8 + r] = (_Float16)(h ? e1[r] : recv);
    }

    // O^T += V^T P^T  (A = V^T dim-chunk x 32 keys, row-contiguous from Vt)
#pragma unroll
    for (int dc = 0; dc < 4; ++dc) {
#pragma unroll
      for (int r = 0; r < 8; ++r) acc[dc][r] *= alpha;
      acc[dc] = WMMA_F16(av[dc], pb, acc[dc]);
    }
  }

  // normalize + store: lane = query, regs = consecutive dims -> b128 stores
  const float inv = 1.0f / lrow;
  float* ob = out + ((size_t)b * SEQ + qrow) * D_K;
#pragma unroll
  for (int dc = 0; dc < 4; ++dc) {
    v4f t0, t1;
#pragma unroll
    for (int i = 0; i < 4; ++i) { t0[i] = acc[dc][i] * inv; t1[i] = acc[dc][4 + i] * inv; }
    *(v4f*)(ob + dc * 16 + 8 * h)     = t0;
    *(v4f*)(ob + dc * 16 + 8 * h + 4) = t1;
  }
}

// ---------------------------------------------------------------------------
extern "C" void kernel_launch(void* const* d_in, const int* in_sizes, int n_in,
                              void* d_out, int out_size, void* d_ws, size_t ws_size,
                              hipStream_t stream) {
  const float* q    = (const float*)d_in[0];
  const float* k    = (const float*)d_in[1];
  const float* v    = (const float*)d_in[2];
  // d_in[3] is the causal mask; causality is computed analytically in-kernel.
  const float* Wq   = (const float*)d_in[4];
  const float* bq   = (const float*)d_in[5];
  const float* Wk   = (const float*)d_in[6];
  const float* bk   = (const float*)d_in[7];
  const float* Wv   = (const float*)d_in[8];
  const float* bv   = (const float*)d_in[9];
  float* outp = (float*)d_out;

  const size_t elems = (size_t)BATCH * SEQ * D_K;   // 1,048,576 f16 each
  _Float16* Qp   = (_Float16*)d_ws;
  _Float16* Kp   = Qp + elems;
  _Float16* Vt   = Kp + elems;
  _Float16* Wswz = Vt + elems;                      // 3 * 64K f16 = 384 KB

  dim3 blk(128);
  wswz_kernel<<<dim3(32, 3), blk, 0, stream>>>(Wq, Wk, Wv, Wswz);

  dim3 pgrid(BATCH * SEQ / 64, 3);   // 256 x 3
  proj_kernel<<<pgrid, blk, 0, stream>>>(q, k, v, bq, bk, bv, Wswz, Qp, Kp, Vt);

  dim3 agrid(SEQ / 64, BATCH);       // 32 x 8
  attn_kernel<<<agrid, blk, 0, stream>>>(Qp, Kp, Vt, outp);
}
